// TriangleMultiplication_30365418782826
// MI455X (gfx1250) — compile-verified, matrix operand order checked by reading, not run on hardware
//
#include <hip/hip_runtime.h>
#include <stdint.h>

// Problem constants (reference: N=512, C_Z=C=128, OUTGOING, EPS=1e-5)
#define NDIM 512
#define CZ   128
#define CC   128
#define NN   (512 * 512)      // 262144 pair rows
#define EPSF 1e-5f

#define AS1 __attribute__((address_space(1)))
#define AS3 __attribute__((address_space(3)))

#if defined(__gfx1250__) && __has_builtin(__builtin_amdgcn_global_load_async_to_lds_b128)
#define HAVE_ASYNC_LDS 1
#else
#define HAVE_ASYNC_LDS 0
#endif

typedef __attribute__((ext_vector_type(16))) __bf16 v16bf;
typedef __attribute__((ext_vector_type(8)))  float  v8f;
typedef __attribute__((ext_vector_type(4)))  int    v4i;

union Frag {
  v16bf v;
  uint4 q[2];
  unsigned short h[16];
};

__device__ __forceinline__ unsigned short f2bf(float f) {
  union { float f; unsigned int u; } cv; cv.f = f;
  unsigned int u = cv.u;
  unsigned int r = (u + 0x7FFFu + ((u >> 16) & 1u)) >> 16;  // round-to-nearest-even
  return (unsigned short)r;
}
__device__ __forceinline__ float bf2f(unsigned short h) {
  union { unsigned int u; float f; } cv; cv.u = ((unsigned int)h) << 16; return cv.f;
}
// fast sigmoid: v_exp_f32 + v_rcp_f32 (avoid precise-division expansion)
__device__ __forceinline__ float sigmoidf_(float x) {
  return __builtin_amdgcn_rcpf(1.0f + __expf(-x));
}
__device__ __forceinline__ float rrsqrt_(float x) {
  return __builtin_amdgcn_rsqf(x);          // v_rsq_f32
}
__device__ __forceinline__ void wait_async0() {
#if HAVE_ASYNC_LDS
#if __has_builtin(__builtin_amdgcn_s_wait_asynccnt)
  __builtin_amdgcn_s_wait_asynccnt(0);
#else
  asm volatile("s_wait_asynccnt 0x0" ::: "memory");
#endif
#endif
}

// ---------------------------------------------------------------------------
// Kernel 0: convert weights fp32 -> bf16, transposed to column-major so the
// WMMA B-operand per-lane load is two contiguous 16B chunks.
// w_cat_t: [640 cols][128 k]  (a1 | a2 | b1 | b2 | g),  w_out_t: [128][128]
// ---------------------------------------------------------------------------
__global__ __launch_bounds__(256) void prep_weights(
    const float* __restrict__ wa1, const float* __restrict__ wa2,
    const float* __restrict__ wb1, const float* __restrict__ wb2,
    const float* __restrict__ wg,  const float* __restrict__ wout,
    unsigned short* __restrict__ w_cat_t, unsigned short* __restrict__ w_out_t) {
  int e = blockIdx.x * blockDim.x + threadIdx.x;   // 768*128 = 98304 total
  if (e >= 768 * 128) return;
  int n = e >> 7, k = e & 127;
  if (n < 640) {
    int blk = n >> 7, col = n & 127;
    const float* src = (blk == 0) ? wa1 : (blk == 1) ? wa2 :
                       (blk == 2) ? wb1 : (blk == 3) ? wb2 : wg;
    w_cat_t[(size_t)n * 128 + k] = f2bf(src[k * 128 + col]);
  } else {
    int col = n - 640;
    w_out_t[(size_t)col * 128 + k] = f2bf(wout[k * 128 + col]);
  }
}

// ---------------------------------------------------------------------------
// Kernel 1: fused LayerNorm(z) + 5 projections + gating epilogue.
// Per block: 128 pair-rows (one i, contiguous j range). 8 waves, wave w owns
// rows [16w,16w+16). K=128 -> 4 WMMA k-steps. 40 output col-tiles.
//   a_t,b_t : bf16, channel-major [c][m]   (m = i*512 + k_pair)
//   g_out   : bf16, row-major     [m][cz]
// ---------------------------------------------------------------------------
__global__ __launch_bounds__(256) void k_lnproj(
    const float* __restrict__ z, const float* __restrict__ smask,
    const float* __restrict__ gam, const float* __restrict__ bet,
    const unsigned short* __restrict__ w_cat_t,
    unsigned short* __restrict__ a_t, unsigned short* __restrict__ b_t,
    unsigned short* __restrict__ g_out) {
  __shared__ unsigned short zn_lds[128][136];   // bf16 zn tile, padded
  __shared__ unsigned short stage[8][256];      // per-wave 16x16 transpose

  const int t = threadIdx.x;
  const size_t m0 = (size_t)blockIdx.x * 128;

  // ---- LayerNorm phase: 2 threads per row (64 channels each) ----
  {
    int row = t >> 1, half = t & 1;
    const float* zr = z + (m0 + row) * (size_t)CZ + half * 64;
    float4 zv[16];
    float s = 0.f, s2 = 0.f;
#pragma unroll
    for (int q = 0; q < 16; ++q) {
      zv[q] = ((const float4*)zr)[q];
      s  += zv[q].x + zv[q].y + zv[q].z + zv[q].w;
      s2 += zv[q].x * zv[q].x + zv[q].y * zv[q].y +
            zv[q].z * zv[q].z + zv[q].w * zv[q].w;
    }
    s  += __shfl_xor(s, 1, 32);    // partner thread covers the other 64 chans
    s2 += __shfl_xor(s2, 1, 32);
    float mean = s * (1.0f / 128.0f);
    float var  = s2 * (1.0f / 128.0f) - mean * mean;
    float rstd = rrsqrt_(var + EPSF);
#pragma unroll
    for (int q = 0; q < 16; ++q) {
      float4 gq = ((const float4*)(gam + half * 64))[q];
      float4 bq = ((const float4*)(bet + half * 64))[q];
      int c0 = half * 64 + q * 4;
      zn_lds[row][c0 + 0] = f2bf((zv[q].x - mean) * rstd * gq.x + bq.x);
      zn_lds[row][c0 + 1] = f2bf((zv[q].y - mean) * rstd * gq.y + bq.y);
      zn_lds[row][c0 + 2] = f2bf((zv[q].z - mean) * rstd * gq.z + bq.z);
      zn_lds[row][c0 + 3] = f2bf((zv[q].w - mean) * rstd * gq.w + bq.w);
    }
  }
  __syncthreads();

  // ---- WMMA GEMM phase ----
  const int wave = t >> 5, lane = t & 31;
  const int lhalf = lane >> 4, l15 = lane & 15;

  // A fragments from LDS (ISA layout: lane row = l15; K chunks {0..7,16..23}
  // for lanes 0-15, {8..15,24..31} for lanes 16-31)
  Frag A[4];
#pragma unroll
  for (int ks = 0; ks < 4; ++ks) {
    const unsigned short* rp = &zn_lds[wave * 16 + l15][0];
    int base = ks * 32 + lhalf * 8;
    A[ks].q[0] = *(const uint4*)(rp + base);
    A[ks].q[1] = *(const uint4*)(rp + base + 16);
  }

  // per-element pair mask (D layout: VGPR v -> M = v + lhalf*8)
  const int ib = (int)(m0 >> 9);
  const int jb = (int)(m0 & 511);
  const float mi = smask[ib];
  float rm[8];
#pragma unroll
  for (int v = 0; v < 8; ++v)
    rm[v] = mi * smask[jb + wave * 16 + lhalf * 8 + v];

  auto compute_tile = [&](int colblk) -> v8f {
    const unsigned short* wrow = w_cat_t + (size_t)(colblk * 16 + l15) * 128;
    v8f acc = {};
#pragma unroll
    for (int ks = 0; ks < 4; ++ks) {
      Frag B;
      int base = ks * 32 + lhalf * 8;
      B.q[0] = *(const uint4*)(wrow + base);
      B.q[1] = *(const uint4*)(wrow + base + 16);
      acc = __builtin_amdgcn_wmma_f32_16x16x32_bf16(
          false, A[ks].v, false, B.v, (short)0, acc, false, false);
    }
    return acc;
  };

  // transpose 16x16 bf16 tile through per-wave LDS -> 16B coalesced stores
  // into channel-major layout dst[(c16+c)*NN + m]
  auto store_ct = [&](unsigned short* dstp, uint4 packed, int c16) {
    unsigned short* sp = &stage[wave][0];
    *(uint4*)(sp + l15 * 16 + lhalf * 8) = packed;       // [c][m-chunk]
    asm volatile("s_wait_dscnt 0x0" ::: "memory");       // cross-lane fence
    int cl = lane >> 1, mh = lane & 1;
    uint4 d = *(const uint4*)(sp + cl * 16 + mh * 8);
    unsigned short* gp =
        dstp + (size_t)(c16 + cl) * NN + m0 + wave * 16 + mh * 8;
    *(uint4*)gp = d;
  };

  // a = sigmoid(zn@w_a1) * (zn@w_a2) * mask  -> a_t (channel-major bf16)
#pragma unroll 1
  for (int tt = 0; tt < 8; ++tt) {
    v8f acc1 = compute_tile(tt);        // a1 cols
    v8f acc2 = compute_tile(8 + tt);    // a2 cols
    union { uint4 q; unsigned short h[8]; } ov;
#pragma unroll
    for (int v = 0; v < 8; ++v)
      ov.h[v] = f2bf(sigmoidf_(acc1[v]) * acc2[v] * rm[v]);
    store_ct(a_t, ov.q, tt * 16);
  }
  // b likewise -> b_t
#pragma unroll 1
  for (int tt = 0; tt < 8; ++tt) {
    v8f acc1 = compute_tile(16 + tt);
    v8f acc2 = compute_tile(24 + tt);
    union { uint4 q; unsigned short h[8]; } ov;
#pragma unroll
    for (int v = 0; v < 8; ++v)
      ov.h[v] = f2bf(sigmoidf_(acc1[v]) * acc2[v] * rm[v]);
    store_ct(b_t, ov.q, tt * 16);
  }
  // g = sigmoid(zn@w_g) -> row-major bf16 [m][cz]
#pragma unroll 1
  for (int tt = 0; tt < 8; ++tt) {
    v8f accg = compute_tile(32 + tt);
#pragma unroll
    for (int v = 0; v < 8; ++v) {
      size_t mrow = m0 + wave * 16 + lhalf * 8 + v;
      g_out[mrow * 128 + tt * 16 + l15] = f2bf(sigmoidf_(accg[v]));
    }
  }
}

// ---------------------------------------------------------------------------
// Kernel 2: triangle einsum. o[i,j,c] = sum_k a[i,k,c] b[j,k,c].
// Per block: one channel c, one 128x128 (i x j) output tile, K=512.
// B-tiles double-buffered in LDS via GLOBAL_LOAD_ASYNC_TO_LDS_B128
// (ASYNCcnt-tracked, no VGPR staging); chunk k+1 loads overlap chunk-k WMMAs.
// A read direct (row per lane). 128 WMMAs per wave. o_t channel-major f32.
// ---------------------------------------------------------------------------
__global__ __launch_bounds__(256) void k_tri(
    const unsigned short* __restrict__ a_t,
    const unsigned short* __restrict__ b_t,
    float* __restrict__ o_t) {
  __shared__ unsigned short bstage[2][128][40];   // 2 x (128 j x 32 k), padded

  const int bid = blockIdx.x;                  // 128 ch * 4 * 4 = 2048
  const int c  = bid >> 4;
  const int i0 = ((bid >> 2) & 3) * 128;
  const int j0 = (bid & 3) * 128;
  const size_t cb = (size_t)c * NN;

  const int t = threadIdx.x, wave = t >> 5, lane = t & 31;
  const int lhalf = lane >> 4, l15 = lane & 15;

  // each thread stages 32B of the B tile per chunk
  const int sj = t >> 1, sko = (t & 1) * 16;
  const unsigned short* bsrc = b_t + cb + (size_t)(j0 + sj) * 512 + sko;

  auto stage_issue = [&](int buf, int kc) {
    const unsigned short* bp = bsrc + kc * 32;
    unsigned short* lp = &bstage[buf][sj][sko];
#if HAVE_ASYNC_LDS
    __builtin_amdgcn_global_load_async_to_lds_b128(
        (AS1 v4i*)bp, (AS3 v4i*)lp, 0, 0);
    __builtin_amdgcn_global_load_async_to_lds_b128(
        (AS1 v4i*)bp, (AS3 v4i*)lp, 16, 0);   // IOFFSET applies to both sides
#else
    *(uint4*)lp       = *(const uint4*)bp;
    *(uint4*)(lp + 8) = *(const uint4*)(bp + 8);
#endif
  };

  v8f acc[8];
#pragma unroll
  for (int x = 0; x < 8; ++x) acc[x] = v8f{};

  const unsigned short* arow =
      a_t + cb + (size_t)(i0 + wave * 16 + l15) * 512;

  stage_issue(0, 0);
  for (int kc = 0; kc < 16; ++kc) {
    wait_async0();                       // this wave's chunk-kc data landed
    if (kc + 1 < 16) stage_issue((kc + 1) & 1, kc + 1);  // prefetch next
    __syncthreads();                     // all waves' chunk-kc data visible

    Frag A;
    {
      int base = kc * 32 + lhalf * 8;
      A.q[0] = *(const uint4*)(arow + base);
      A.q[1] = *(const uint4*)(arow + base + 16);
    }
#pragma unroll
    for (int tj = 0; tj < 8; ++tj) {
      Frag B;
      const unsigned short* rp = &bstage[kc & 1][tj * 16 + l15][0];
      B.q[0] = *(const uint4*)(rp + lhalf * 8);
      B.q[1] = *(const uint4*)(rp + lhalf * 8 + 16);
      acc[tj] = __builtin_amdgcn_wmma_f32_16x16x32_bf16(
          false, A.v, false, B.v, (short)0, acc[tj], false, false);
    }
    __syncthreads();                     // done reading before buffer reuse
  }

#pragma unroll 1
  for (int tj = 0; tj < 8; ++tj) {
#pragma unroll
    for (int v = 0; v < 8; ++v) {
      int gi = i0 + wave * 16 + lhalf * 8 + v;
      int gj = j0 + tj * 16 + l15;
      o_t[cb + (size_t)gi * 512 + gj] = acc[tj][v];
    }
  }
}

// ---------------------------------------------------------------------------
// Kernel 3: out = g * (LayerNorm_c(o) @ w_out).
// Per block: 128 pair-rows. o is channel-major so per-channel gathers are
// lane-coalesced. Cross-wave LN reduction via LDS, then WMMA with w_out^T.
// ---------------------------------------------------------------------------
__global__ __launch_bounds__(256) void k_out(
    const float* __restrict__ o_t, const unsigned short* __restrict__ g_in,
    const float* __restrict__ gam, const float* __restrict__ bet,
    const unsigned short* __restrict__ w_out_t,
    float* __restrict__ out) {
  __shared__ unsigned short on_lds[128][136];
  __shared__ float psum[256], psq[256];

  const int t = threadIdx.x;
  const size_t m0 = (size_t)blockIdx.x * 128;

  { // LN over channels: 2 threads per row m, 64 channels each
    int ml = t & 127, ch = t >> 7;
    float vals[64];
    float s = 0.f, s2 = 0.f;
#pragma unroll
    for (int q = 0; q < 64; ++q) {
      float v = o_t[(size_t)(ch * 64 + q) * NN + m0 + ml];
      vals[q] = v; s += v; s2 += v * v;
    }
    psum[t] = s; psq[t] = s2;
    __syncthreads();
    float S  = psum[ml] + psum[ml + 128];
    float S2 = psq[ml] + psq[ml + 128];
    float mean = S * (1.0f / 128.0f);
    float var  = S2 * (1.0f / 128.0f) - mean * mean;
    float rstd = rrsqrt_(var + EPSF);
#pragma unroll
    for (int q = 0; q < 64; ++q) {
      int cc = ch * 64 + q;
      on_lds[ml][cc] = f2bf((vals[q] - mean) * rstd * gam[cc] + bet[cc]);
    }
  }
  __syncthreads();

  const int wave = t >> 5, lane = t & 31;
  const int lhalf = lane >> 4, l15 = lane & 15;

  Frag A[4];
#pragma unroll
  for (int ks = 0; ks < 4; ++ks) {
    const unsigned short* rp = &on_lds[wave * 16 + l15][0];
    int base = ks * 32 + lhalf * 8;
    A[ks].q[0] = *(const uint4*)(rp + base);
    A[ks].q[1] = *(const uint4*)(rp + base + 16);
  }

#pragma unroll 1
  for (int nt = 0; nt < 8; ++nt) {
    const unsigned short* wrow = w_out_t + (size_t)(nt * 16 + l15) * 128;
    v8f acc = {};
#pragma unroll
    for (int ks = 0; ks < 4; ++ks) {
      Frag B;
      int base = ks * 32 + lhalf * 8;
      B.q[0] = *(const uint4*)(wrow + base);
      B.q[1] = *(const uint4*)(wrow + base + 16);
      acc = __builtin_amdgcn_wmma_f32_16x16x32_bf16(
          false, A[ks].v, false, B.v, (short)0, acc, false, false);
    }
#pragma unroll
    for (int v = 0; v < 8; ++v) {
      size_t mrow = m0 + wave * 16 + lhalf * 8 + v;
      int col = nt * 16 + l15;
      float gv = bf2f(g_in[mrow * 128 + col]);
      out[mrow * 128 + col] = gv * acc[v];
    }
  }
}

// ---------------------------------------------------------------------------
extern "C" void kernel_launch(void* const* d_in, const int* in_sizes, int n_in,
                              void* d_out, int out_size, void* d_ws, size_t ws_size,
                              hipStream_t stream) {
  const float* z     = (const float*)d_in[0];
  const float* smask = (const float*)d_in[1];
  const float* lnzg  = (const float*)d_in[2];
  const float* lnzb  = (const float*)d_in[3];
  const float* wa1   = (const float*)d_in[4];
  const float* wa2   = (const float*)d_in[5];
  const float* wb1   = (const float*)d_in[6];
  const float* wb2   = (const float*)d_in[7];
  const float* wg    = (const float*)d_in[8];
  const float* lnog  = (const float*)d_in[9];
  const float* lnob  = (const float*)d_in[10];
  const float* wout  = (const float*)d_in[11];

  // workspace layout
  char* p = (char*)d_ws;
  size_t off = 0;
  unsigned short* a_t = (unsigned short*)(p + off); off += (size_t)CC * NN * 2; // 64 MB
  unsigned short* b_t = (unsigned short*)(p + off); off += (size_t)CC * NN * 2; // 64 MB
  unsigned short* g_b = (unsigned short*)(p + off); off += (size_t)NN * CZ * 2; // 64 MB
  float*          o_t = (float*)(p + off);          off += (size_t)CC * NN * 4; // 128 MB
  unsigned short* wct = (unsigned short*)(p + off); off += (size_t)640 * 128 * 2;
  unsigned short* wot = (unsigned short*)(p + off); off += (size_t)128 * 128 * 2;
  if (off > ws_size) return;  // insufficient scratch

  prep_weights<<<384, 256, 0, stream>>>(wa1, wa2, wb1, wb2, wg, wout, wct, wot);
  k_lnproj<<<2048, 256, 0, stream>>>(z, smask, lnzg, lnzb, wct, a_t, b_t, g_b);
  k_tri<<<2048, 256, 0, stream>>>(a_t, b_t, o_t);
  k_out<<<2048, 256, 0, stream>>>(o_t, g_b, lnog, lnob, wot, (float*)d_out);
}
